// Model_30374008717907
// MI455X (gfx1250) — matrix operand order repaired and checked by reading, not compiled, not run on hardware
//
#include <hip/hip_runtime.h>
#include <hip/hip_bf16.h>
#include <math.h>

typedef __attribute__((ext_vector_type(16))) _Float16 v16h;
typedef __attribute__((ext_vector_type(8)))  _Float16 v8h;
typedef __attribute__((ext_vector_type(8)))  float    v8f;
typedef int v4i_vs __attribute__((vector_size(16)));  // matches builtin param

// Problem constants
static constexpr int kB = 32, kT = 512, kC = 32;
static constexpr int kD = 512, kH = 8, kDFF = 2048, kNL = 3;
static constexpr int kP = 16, kN = 64, kE = 64;
static constexpr int kSeqs = kB * kC;          // 1024
static constexpr int kM = kSeqs * kN;          // 65536 token rows

// ---------------------------------------------------------------------------
// gfx1250 async global->LDS staging (ASYNCcnt path) with sync fallback
// ---------------------------------------------------------------------------
#if __has_builtin(__builtin_amdgcn_global_load_async_to_lds_b128) && \
    __has_builtin(__builtin_amdgcn_s_wait_asynccnt)
#define HAVE_ASYNC_LDS 1
__device__ __forceinline__ void async_cp16(void* lds, const void* g) {
  __builtin_amdgcn_global_load_async_to_lds_b128(
      (__attribute__((address_space(1))) v4i_vs*)(uintptr_t)g,
      (__attribute__((address_space(3))) v4i_vs*)(uint32_t)(uintptr_t)lds,
      0, 0);
}
__device__ __forceinline__ void async_wait() {
  __builtin_amdgcn_s_wait_asynccnt(0);
}
#else
#define HAVE_ASYNC_LDS 0
__device__ __forceinline__ void async_cp16(void* lds, const void* g) {
  *(v8h*)lds = *(const v8h*)g;
}
__device__ __forceinline__ void async_wait() {}
#endif

// xor-lane reduction step via one ds_swizzle_b32 (masks <16 stay in-half)
template <int MASK>
__device__ __forceinline__ float swz_xor(float x) {
  return __int_as_float(__builtin_amdgcn_ds_swizzle(
      __float_as_int(x), (MASK << 10) | 0x1f));
}

// ---------------------------------------------------------------------------
// Generic WMMA GEMM: C[M,Nc] = A[M,K](f16,row-major) * B[K,Nc]
// B is supplied PRE-TRANSPOSED: BwT[Nc][K] row-major (weight repack kernel).
// FLAGS: 1=+bias(col) 2=+residual(f32) 4=GELU 8=store f16 16=store f32
// Block: 256 threads (8 waves). Tile: 256 rows x 64 cols, K-step 32,
// double-buffered async global->LDS staging; wave = 32-row strip.
// ---------------------------------------------------------------------------
template <int FLAGS>
__global__ __launch_bounds__(256) void gemm_wmma(
    const _Float16* __restrict__ A, const _Float16* __restrict__ BwT,
    const float* __restrict__ bias, const float* __restrict__ res,
    float* __restrict__ out32, _Float16* __restrict__ out16,
    int M, int Ncols, int K) {
  __shared__ __align__(16) _Float16 As[2][256][32];
  __shared__ __align__(16) _Float16 Bt[2][64][32];  // Bt[buf][n][k]

  const int tid = threadIdx.x;
  const int wave = tid >> 5;
  const int lane = tid & 31;
  const int lrow = lane & 15;
  const int lhi = lane >> 4;
  const int rowbase = blockIdx.x * 256;
  const int colbase = blockIdx.y * 64;

  // staging coordinates (coalesced 16B chunks)
  const int ar = tid >> 1;           // 0..127
  const int ako = (tid & 1) * 16;
  const int bn = tid >> 2;
  const int bk0 = (tid & 3) * 8;
  const _Float16* srcA0 = A + (size_t)(rowbase + ar) * K + ako;
  const _Float16* srcA1 = A + (size_t)(rowbase + 128 + ar) * K + ako;
  const _Float16* srcB0 = BwT + (size_t)(colbase + bn) * K + bk0;

  auto stage = [&](int buf, int kb) {
    async_cp16(&As[buf][ar][ako], srcA0 + kb);
    async_cp16(&As[buf][ar][ako + 8], srcA0 + kb + 8);
    async_cp16(&As[buf][128 + ar][ako], srcA1 + kb);
    async_cp16(&As[buf][128 + ar][ako + 8], srcA1 + kb + 8);
    async_cp16(&Bt[buf][bn][bk0], srcB0 + kb);
  };

  v8f acc[2][4];
  v8f vzero = {};
#pragma unroll
  for (int r = 0; r < 2; ++r)
#pragma unroll
    for (int c = 0; c < 4; ++c) acc[r][c] = vzero;

  stage(0, 0);
  async_wait();
  __syncthreads();

  int cur = 0;
  for (int kb = 0; kb < K; kb += 32) {
    // prefetch next K-step into the other buffer while computing this one
    if (kb + 32 < K) stage(cur ^ 1, kb + 32);

    const int s8 = lhi * 8;
    const int kb16 = lhi * 16;
    v16h afrag[2];
#pragma unroll
    for (int r = 0; r < 2; ++r) {
      const int m = wave * 32 + r * 16 + lrow;
      v8h alo = *(const v8h*)&As[cur][m][s8];
      v8h ahi = *(const v8h*)&As[cur][m][16 + s8];
      afrag[r] = __builtin_shufflevector(alo, ahi, 0, 1, 2, 3, 4, 5, 6, 7, 8,
                                         9, 10, 11, 12, 13, 14, 15);
    }
#pragma unroll
    for (int c = 0; c < 4; ++c) {
      const int n = c * 16 + lrow;
      v8h blo = *(const v8h*)&Bt[cur][n][s8];
      v8h bhi = *(const v8h*)&Bt[cur][n][16 + s8];
      v16h bfrag = __builtin_shufflevector(blo, bhi, 0, 1, 2, 3, 4, 5, 6, 7, 8,
                                           9, 10, 11, 12, 13, 14, 15);
#pragma unroll
      for (int r = 0; r < 2; ++r) {
        acc[r][c] = __builtin_amdgcn_wmma_f32_16x16x32_f16(
            false, afrag[r], false, bfrag, (short)0, acc[r][c], false, false);
      }
    }

    async_wait();     // my async stores into buf cur^1 have landed
    __syncthreads();  // all waves done reading buf cur
    cur ^= 1;
  }

  // --- epilogue ---
#pragma unroll
  for (int r = 0; r < 2; ++r) {
#pragma unroll
    for (int c = 0; c < 4; ++c) {
      const int col = colbase + c * 16 + lrow;
      const float bv = (FLAGS & 1) ? bias[col] : 0.0f;
#pragma unroll
      for (int v = 0; v < 8; ++v) {
        const int row = rowbase + wave * 32 + r * 16 + v + 8 * lhi;
        const size_t idx = (size_t)row * Ncols + col;
        float x = acc[r][c][v] + bv;
        if (FLAGS & 2) x += res[idx];
        if (FLAGS & 4) x = 0.5f * x * (1.0f + erff(x * 0.70710678118654752f));
        if (FLAGS & 16) out32[idx] = x;
        if (FLAGS & 8) out16[idx] = (_Float16)x;
      }
    }
  }
}

// ---------------------------------------------------------------------------
// Attention: one block per (seq, 16-row strip); wave w = head w.
// q/k layout: [seq*64 + tok][512] f16, head h at cols h*64..h*64+63
// vT layout:  [((seq*8 + h)*64 + e)][64 tokens]  (pre-transposed)
// ---------------------------------------------------------------------------
__global__ __launch_bounds__(256) void attention_kernel(
    const _Float16* __restrict__ q, const _Float16* __restrict__ k,
    const _Float16* __restrict__ vT, _Float16* __restrict__ o) {
  __shared__ __align__(16) _Float16 Ps[8][16][64];
  const int tid = threadIdx.x;
  const int h = tid >> 5;  // wave == head
  const int lane = tid & 31;
  const int lrow = lane & 15;
  const int lhi = lane >> 4;
  const int seq = blockIdx.x >> 2;
  const int strip = blockIdx.x & 3;
  const size_t base = (size_t)seq * 64 * kD + h * kE;
  const size_t vbase = ((size_t)seq * kH + h) * kE * 64;

  v8f vzero = {};
  v8f sc[4];
#pragma unroll
  for (int c = 0; c < 4; ++c) sc[c] = vzero;

  // scores: A = q[16 x 64], B = k^T (K=e, N=token)
#pragma unroll
  for (int ks = 0; ks < kE; ks += 32) {
    const int m = strip * 16 + lrow;
    const int s8 = lhi * 8;
    const _Float16* ap = q + base + (size_t)m * kD + ks + s8;
    v8h alo = *(const v8h*)ap;
    v8h ahi = *(const v8h*)(ap + 16);
    v16h af = __builtin_shufflevector(alo, ahi, 0, 1, 2, 3, 4, 5, 6, 7, 8, 9,
                                      10, 11, 12, 13, 14, 15);
#pragma unroll
    for (int c = 0; c < 4; ++c) {
      const int s = c * 16 + lrow;  // key token
      const _Float16* bp = k + base + (size_t)s * kD + ks + lhi * 8;
      v8h blo = *(const v8h*)bp;
      v8h bhi = *(const v8h*)(bp + 16);
      v16h bf = __builtin_shufflevector(blo, bhi, 0, 1, 2, 3, 4, 5, 6, 7, 8, 9,
                                        10, 11, 12, 13, 14, 15);
      sc[c] = __builtin_amdgcn_wmma_f32_16x16x32_f16(false, af, false, bf,
                                                     (short)0, sc[c], false,
                                                     false);
    }
  }

  // softmax per row (row = v + 8*lhi, cols over lanes 0..15 x 4 tiles)
  const float scale = 0.125f;  // 1/sqrt(64)
#pragma unroll
  for (int vv = 0; vv < 8; ++vv) {
    float t0 = scale * sc[0][vv], t1 = scale * sc[1][vv];
    float t2 = scale * sc[2][vv], t3 = scale * sc[3][vv];
    float mx = fmaxf(fmaxf(t0, t1), fmaxf(t2, t3));
    mx = fmaxf(mx, swz_xor<1>(mx));
    mx = fmaxf(mx, swz_xor<2>(mx));
    mx = fmaxf(mx, swz_xor<4>(mx));
    mx = fmaxf(mx, swz_xor<8>(mx));
    float e0 = __expf(t0 - mx), e1 = __expf(t1 - mx);
    float e2 = __expf(t2 - mx), e3 = __expf(t3 - mx);
    float sum = e0 + e1 + e2 + e3;
    sum += swz_xor<1>(sum);
    sum += swz_xor<2>(sum);
    sum += swz_xor<4>(sum);
    sum += swz_xor<8>(sum);
    const float inv = 1.0f / sum;
    const int rloc = vv + 8 * lhi;
    Ps[h][rloc][0 * 16 + lrow] = (_Float16)(e0 * inv);
    Ps[h][rloc][1 * 16 + lrow] = (_Float16)(e1 * inv);
    Ps[h][rloc][2 * 16 + lrow] = (_Float16)(e2 * inv);
    Ps[h][rloc][3 * 16 + lrow] = (_Float16)(e3 * inv);
  }
  __syncthreads();

  // out = P @ V : A = Ps[16 x 64], B = v (K=token, N=e) via vT
  v8f ov[4];
#pragma unroll
  for (int c = 0; c < 4; ++c) ov[c] = vzero;
#pragma unroll
  for (int ks = 0; ks < 64; ks += 32) {
    const int s8 = lhi * 8;
    v8h alo = *(const v8h*)&Ps[h][lrow][ks + s8];
    v8h ahi = *(const v8h*)&Ps[h][lrow][ks + 16 + s8];
    v16h af = __builtin_shufflevector(alo, ahi, 0, 1, 2, 3, 4, 5, 6, 7, 8, 9,
                                      10, 11, 12, 13, 14, 15);
#pragma unroll
    for (int c = 0; c < 4; ++c) {
      const int e = c * 16 + lrow;
      const int k0 = ks + lhi * 8;
      const _Float16* bp = vT + vbase + (size_t)e * 64 + k0;
      v8h blo = *(const v8h*)bp;
      v8h bhi = *(const v8h*)(bp + 16);
      v16h bf = __builtin_shufflevector(blo, bhi, 0, 1, 2, 3, 4, 5, 6, 7, 8, 9,
                                        10, 11, 12, 13, 14, 15);
      ov[c] = __builtin_amdgcn_wmma_f32_16x16x32_f16(false, af, false, bf,
                                                     (short)0, ov[c], false,
                                                     false);
    }
  }

#pragma unroll
  for (int c = 0; c < 4; ++c) {
    const int e = c * 16 + lrow;
#pragma unroll
    for (int vv = 0; vv < 8; ++vv) {
      const int row = strip * 16 + vv + 8 * lhi;
      o[base + (size_t)row * kD + e] = (_Float16)ov[c][vv];
    }
  }
}

// ---------------------------------------------------------------------------
// Small VALU kernels
// ---------------------------------------------------------------------------
__global__ __launch_bounds__(256) void instnorm_stats(
    const float* __restrict__ x, float* __restrict__ mean,
    float* __restrict__ stdev) {
  __shared__ float s1[256], s2[256];
  const int tid = threadIdx.x;
  const int m = blockIdx.x;
  const int b = m >> 5, c = m & 31;
  float s = 0.f, ss = 0.f;
  for (int t = tid; t < kT; t += 256) {
    float v = x[((size_t)b * kT + t) * kC + c];
    s += v;
    ss += v * v;
  }
  s1[tid] = s;
  s2[tid] = ss;
  __syncthreads();
  for (int o = 128; o > 0; o >>= 1) {
    if (tid < o) { s1[tid] += s1[tid + o]; s2[tid] += s2[tid + o]; }
    __syncthreads();
  }
  if (tid == 0) {
    float mu = s1[0] / (float)kT;
    float var = s2[0] / (float)kT - mu * mu;
    mean[m] = mu;
    stdev[m] = sqrtf(var + 1e-5f);
  }
}

__global__ __launch_bounds__(128) void patch_embed(
    const float* __restrict__ x, const float* __restrict__ Wval,
    const float* __restrict__ mean, const float* __restrict__ stdev,
    float* __restrict__ tok32, _Float16* __restrict__ tok16) {
  __shared__ float pv[16];
  const int blk = blockIdx.x;
  const int m = blk >> 6;   // seq
  const int n = blk & 63;   // patch
  const int b = m >> 5, c = m & 31;
  const int tid = threadIdx.x;
  if (tid < 16) {
    int t = n * 8 + tid;
    if (t > kT - 1) t = kT - 1;  // replication pad
    pv[tid] = (x[((size_t)b * kT + t) * kC + c] - mean[m]) / stdev[m];
  }
  __syncthreads();
  for (int d = tid; d < kD; d += 128) {
    float div = __expf(-(float)((d >> 1) << 1) * (9.210340371976184f / (float)kD));
    float ang = (float)n * div;
    float acc = (d & 1) ? __cosf(ang) : __sinf(ang);
#pragma unroll
    for (int p = 0; p < 16; ++p) acc += pv[p] * Wval[p * kD + d];
    size_t idx = (size_t)(m * kN + n) * kD + d;
    tok32[idx] = acc;
    tok16[idx] = (_Float16)acc;
  }
}

__global__ __launch_bounds__(256) void layernorm_kernel(
    const float* __restrict__ x, const float* __restrict__ g,
    const float* __restrict__ bta, float* __restrict__ out32,
    _Float16* __restrict__ out16) {
  __shared__ float s1[256], s2[256];
  const int tid = threadIdx.x;
  const size_t row = blockIdx.x;
  const float* xr = x + row * kD;
  float a = xr[tid], b2 = xr[tid + 256];
  s1[tid] = a + b2;
  s2[tid] = a * a + b2 * b2;
  __syncthreads();
  for (int o = 128; o > 0; o >>= 1) {
    if (tid < o) { s1[tid] += s1[tid + o]; s2[tid] += s2[tid + o]; }
    __syncthreads();
  }
  const float mu = s1[0] / (float)kD;
  const float var = s2[0] / (float)kD - mu * mu;
  const float rstd = rsqrtf(var + 1e-5f);
#pragma unroll
  for (int dd = 0; dd < 2; ++dd) {
    const int d = tid + dd * 256;
    float v = (xr[d] - mu) * rstd * g[d] + bta[d];
    out16[row * kD + d] = (_Float16)v;
    if (out32) out32[row * kD + d] = v;
  }
}

__global__ __launch_bounds__(256) void bn_stats(const float* __restrict__ tok,
                                                float* __restrict__ bnm,
                                                float* __restrict__ bnr) {
  __shared__ float s1[256], s2[256];
  const int tid = threadIdx.x;
  const int d = blockIdx.x;
  float s = 0.f, ss = 0.f;
  for (int r = tid; r < kM; r += 256) {
    float v = tok[(size_t)r * kD + d];
    s += v;
    ss += v * v;
  }
  s1[tid] = s;
  s2[tid] = ss;
  __syncthreads();
  for (int o = 128; o > 0; o >>= 1) {
    if (tid < o) { s1[tid] += s1[tid + o]; s2[tid] += s2[tid + o]; }
    __syncthreads();
  }
  if (tid == 0) {
    float mu = s1[0] / (float)kM;
    float var = s2[0] / (float)kM - mu * mu;
    bnm[d] = mu;
    bnr[d] = rsqrtf(var + 1e-5f);
  }
}

// BN apply + transpose into head-GEMM A layout: headA[seq][d*64+n]
__global__ __launch_bounds__(256) void bn_pack(
    const float* __restrict__ tok, const float* __restrict__ bnm,
    const float* __restrict__ bnr, const float* __restrict__ g,
    const float* __restrict__ bb, _Float16* __restrict__ headA) {
  size_t idx = (size_t)blockIdx.x * 256 + threadIdx.x;
  if (idx >= (size_t)kM * kD) return;
  const int d = idx & (kD - 1);
  const size_t m = idx >> 9;
  const int n = (int)(m & 63);
  const size_t seq = m >> 6;
  float v = (tok[idx] - bnm[d]) * bnr[d] * g[d] + bb[d];
  headA[seq * (size_t)(kD * kN) + (size_t)d * kN + n] = (_Float16)v;
}

// out[b,t,c] = dec[b*32+c][t] * stdev + mean
__global__ __launch_bounds__(256) void final_out(
    const float* __restrict__ dec, const float* __restrict__ mean,
    const float* __restrict__ stdev, float* __restrict__ out) {
  const int idx = blockIdx.x * 256 + threadIdx.x;
  if (idx >= kB * kT * kC) return;
  const int c = idx & 31;
  const int t = (idx >> 5) & (kT - 1);
  const int b = idx >> 14;
  const int m = b * kC + c;
  out[idx] = dec[(size_t)m * kT + t] * stdev[m] + mean[m];
}

// f32 -> f16 with per-matrix transpose: out[mat][n][k] = in[mat][k][n]
__global__ __launch_bounds__(256) void f32_to_f16_T(
    const float* __restrict__ in, _Float16* __restrict__ out, int K, int Nc,
    int nmats) {
  size_t idx = (size_t)blockIdx.x * 256 + threadIdx.x;
  const size_t per = (size_t)K * Nc;
  if (idx >= per * (size_t)nmats) return;
  const size_t mat = idx / per;
  const size_t r = idx - mat * per;
  const int k = (int)(r / Nc);
  const int n = (int)(r - (size_t)k * Nc);
  out[mat * per + (size_t)n * K + k] = (_Float16)in[idx];
}

// v [seq*64+s][h*64+e] -> vT [((seq*8+h)*64+e)][s]
__global__ __launch_bounds__(256) void transpose_v(
    const _Float16* __restrict__ v, _Float16* __restrict__ vT) {
  size_t idx = (size_t)blockIdx.x * 256 + threadIdx.x;
  if (idx >= (size_t)kM * kD) return;
  const int col = idx & (kD - 1);
  const size_t row = idx >> 9;
  const int s = (int)(row & 63);
  const size_t seq = row >> 6;
  const int h = col >> 6;
  const int e = col & 63;
  vT[(((size_t)seq * kH + h) * kE + e) * 64 + s] = v[idx];
}

// ---------------------------------------------------------------------------
extern "C" void kernel_launch(void* const* d_in, const int* in_sizes, int n_in,
                              void* d_out, int out_size, void* d_ws,
                              size_t ws_size, hipStream_t stream) {
  const float* x_enc = (const float*)d_in[0];
  const float* W_val = (const float*)d_in[1];
  const float* Wq = (const float*)d_in[2];
  const float* bq = (const float*)d_in[3];
  const float* Wk = (const float*)d_in[4];
  const float* bk = (const float*)d_in[5];
  const float* Wv = (const float*)d_in[6];
  const float* bv = (const float*)d_in[7];
  const float* Wo = (const float*)d_in[8];
  const float* bo = (const float*)d_in[9];
  const float* Wc1 = (const float*)d_in[10];
  const float* bc1 = (const float*)d_in[11];
  const float* Wc2 = (const float*)d_in[12];
  const float* bc2 = (const float*)d_in[13];
  const float* ln1_g = (const float*)d_in[14];
  const float* ln1_b = (const float*)d_in[15];
  const float* ln2_g = (const float*)d_in[16];
  const float* ln2_b = (const float*)d_in[17];
  const float* bn_g = (const float*)d_in[18];
  const float* bn_b = (const float*)d_in[19];
  const float* W_head = (const float*)d_in[20];
  const float* b_head = (const float*)d_in[21];
  float* out = (float*)d_out;

  // workspace allocator (256B aligned)
  char* ws = (char*)d_ws;
  size_t off = 0;
  auto alloc = [&](size_t bytes) -> char* {
    char* p = ws + off;
    off = (off + bytes + 255) & ~(size_t)255;
    return p;
  };

  float* mean_a = (float*)alloc(kSeqs * 4);
  float* std_a = (float*)alloc(kSeqs * 4);
  float* bnm = (float*)alloc(kD * 4);
  float* bnr = (float*)alloc(kD * 4);
  float* tok32 = (float*)alloc((size_t)kM * kD * 4);
  float* pre32 = (float*)alloc((size_t)kM * kD * 4);
  float* dec32 = (float*)alloc((size_t)kSeqs * kT * 4);
  _Float16* tok16 = (_Float16*)alloc((size_t)kM * kD * 2);
  _Float16* q16 = (_Float16*)alloc((size_t)kM * kD * 2);
  _Float16* k16 = (_Float16*)alloc((size_t)kM * kD * 2);
  _Float16* v16 = (_Float16*)alloc((size_t)kM * kD * 2);
  _Float16* vT16 = (_Float16*)alloc((size_t)kM * kD * 2);
  _Float16* at16 = (_Float16*)alloc((size_t)kM * kD * 2);
  _Float16* y16 = (_Float16*)alloc((size_t)kM * kD * 2);
  _Float16* h16 = (_Float16*)alloc((size_t)kM * kDFF * 2);
  _Float16* headA = (_Float16*)alloc((size_t)kSeqs * kD * kN * 2);
  _Float16* wq16 = (_Float16*)alloc((size_t)kNL * kD * kD * 2);
  _Float16* wk16 = (_Float16*)alloc((size_t)kNL * kD * kD * 2);
  _Float16* wv16 = (_Float16*)alloc((size_t)kNL * kD * kD * 2);
  _Float16* wo16 = (_Float16*)alloc((size_t)kNL * kD * kD * 2);
  _Float16* wc116 = (_Float16*)alloc((size_t)kNL * kD * kDFF * 2);
  _Float16* wc216 = (_Float16*)alloc((size_t)kNL * kDFF * kD * 2);
  _Float16* wh16 = (_Float16*)alloc((size_t)kD * kN * kT * 2);

  // weight repack: f32 [K][N] -> f16 transposed [N][K]
  auto cvtT = [&](const float* src, _Float16* dst, int K, int Nc, int nm) {
    size_t n = (size_t)K * Nc * nm;
    f32_to_f16_T<<<(unsigned)((n + 255) / 256), 256, 0, stream>>>(src, dst, K,
                                                                  Nc, nm);
  };
  cvtT(Wq, wq16, kD, kD, kNL);
  cvtT(Wk, wk16, kD, kD, kNL);
  cvtT(Wv, wv16, kD, kD, kNL);
  cvtT(Wo, wo16, kD, kD, kNL);
  cvtT(Wc1, wc116, kD, kDFF, kNL);
  cvtT(Wc2, wc216, kDFF, kD, kNL);
  cvtT(W_head, wh16, kD * kN, kT, 1);

  instnorm_stats<<<kSeqs, 256, 0, stream>>>(x_enc, mean_a, std_a);
  patch_embed<<<kSeqs * kN, 128, 0, stream>>>(x_enc, W_val, mean_a, std_a,
                                              tok32, tok16);

  const dim3 gD(kM / 256, kD / 64);        // (256, 8)
  const dim3 gFF(kM / 256, kDFF / 64);     // (256, 32)
  const unsigned gElem = (unsigned)(((size_t)kM * kD + 255) / 256);
  for (int i = 0; i < kNL; ++i) {
    const size_t wofs = (size_t)i * kD * kD;
    gemm_wmma<9><<<gD, 256, 0, stream>>>(tok16, wq16 + wofs, bq + i * kD,
                                         nullptr, nullptr, q16, kM, kD, kD);
    gemm_wmma<9><<<gD, 256, 0, stream>>>(tok16, wk16 + wofs, bk + i * kD,
                                         nullptr, nullptr, k16, kM, kD, kD);
    gemm_wmma<9><<<gD, 256, 0, stream>>>(tok16, wv16 + wofs, bv + i * kD,
                                         nullptr, nullptr, v16, kM, kD, kD);
    transpose_v<<<gElem, 256, 0, stream>>>(v16, vT16);
    attention_kernel<<<kSeqs * 4, 256, 0, stream>>>(q16, k16, vT16, at16);
    gemm_wmma<19><<<gD, 256, 0, stream>>>(at16, wo16 + wofs, bo + i * kD,
                                          tok32, tok32, nullptr, kM, kD, kD);
    layernorm_kernel<<<kM, 256, 0, stream>>>(tok32, ln1_g + i * kD,
                                             ln1_b + i * kD, nullptr, y16);
    gemm_wmma<13><<<gFF, 256, 0, stream>>>(y16, wc116 + (size_t)i * kD * kDFF,
                                           bc1 + i * kDFF, nullptr, nullptr,
                                           h16, kM, kDFF, kD);
    gemm_wmma<19><<<gD, 256, 0, stream>>>(h16, wc216 + (size_t)i * kDFF * kD,
                                          bc2 + i * kD, tok32, pre32, nullptr,
                                          kM, kD, kDFF);
    layernorm_kernel<<<kM, 256, 0, stream>>>(pre32, ln2_g + i * kD,
                                             ln2_b + i * kD, tok32, tok16);
  }

  bn_stats<<<kD, 256, 0, stream>>>(tok32, bnm, bnr);
  bn_pack<<<gElem, 256, 0, stream>>>(tok32, bnm, bnr, bn_g, bn_b, headA);

  const dim3 gHead(kSeqs / 256, kT / 64);  // (4, 8)
  gemm_wmma<17><<<gHead, 256, 0, stream>>>(headA, wh16, b_head, nullptr, dec32,
                                           nullptr, kSeqs, kT, kD * kN);
  final_out<<<(kB * kT * kC + 255) / 256, 256, 0, stream>>>(dec32, mean_a,
                                                            std_a, out);
}
